// ProductionMoE_1322849927638
// MI455X (gfx1250) — compile-verified, matrix operand order checked by reading, not run on hardware
//
#include <hip/hip_runtime.h>
#include <hip/hip_bf16.h>

// ---------------------------------------------------------------------------
// MoE forward for MI455X (gfx1250), wave32 + v_wmma_f32_16x16x32_bf16.
// Compute-bound (~515 GFLOP vs ~0.5 GB traffic @ 23.3 TB/s): all three expert
// GEMMs run on bf16 WMMA with f32 accumulation. Both top-k dispatches share
// one 2*capacity row block per expert so each weight matrix streams once.
// GEMM2 A-tiles stream via CDNA5 async global->LDS (double buffered).
// Fragments are batch-loaded before each WMMA chain so the scheduler can
// issue one s_wait_dscnt and then back-to-back v_wmma.
// ---------------------------------------------------------------------------

#define MOE_ASYNC 1

typedef __bf16 v16bf __attribute__((ext_vector_type(16)));
typedef float  v8f   __attribute__((ext_vector_type(8)));

static constexpr int BM  = 128;  // workgroup rows (slots)
static constexpr int BN  = 64;   // workgroup cols
static constexpr int BK  = 32;   // GEMM1 K per stage == WMMA K
static constexpr int LS  = 40;   // GEMM1 LDS row stride (bf16 elems, 80B)
static constexpr int BK2 = 64;   // GEMM2 K per stage (2 WMMA k-steps)
static constexpr int LS2 = 72;   // GEMM2 LDS row stride (bf16 elems, 144B)

// Fragment load from an LDS tile stored row-major [row][stride] bf16.
// ISA 7.12.2 bf16 16x32 A layout: lanes 0-15 row m hold K {0..7,16..23},
// lanes 16-31 hold K {8..15,24..31}; applied symmetrically for B.
__device__ __forceinline__ v16bf ldsFrag(const __bf16* base, int lane, int stride) {
  const int r  = lane & 15;
  const int kb = (lane & 16) ? 8 : 0;
  const __bf16* p = base + r * stride + kb;
  v16bf f;
#pragma unroll
  for (int j = 0; j < 8; ++j) f[j] = p[j];
#pragma unroll
  for (int j = 0; j < 8; ++j) f[j + 8] = p[16 + j];
  return f;
}

__device__ __forceinline__ v8f wmma_bf16(v16bf a, v16bf b, v8f c) {
  return __builtin_amdgcn_wmma_f32_16x16x32_bf16(false, a, false, b, (short)0, c,
                                                 false, false);
}

// ---------------------------------------------------------------- utilities
__global__ void moe_zero_out(float* __restrict__ p, long long n) {
  long long i  = (long long)blockIdx.x * blockDim.x + threadIdx.x;
  long long st = (long long)gridDim.x * blockDim.x;
  for (; i < n; i += st) p[i] = 0.0f;
}

__global__ void moe_init_slots(int* __restrict__ s2t, float* __restrict__ swt, int n) {
  int i = blockIdx.x * blockDim.x + threadIdx.x;
  if (i < n) { s2t[i] = -1; swt[i] = 0.0f; }
}

// ------------------------------------------------------------------- router
// One wave per token: lane-strided dot over D, shuffle-reduce, softmax+top2.
__global__ __launch_bounds__(256) void moe_router(const float* __restrict__ x,
                                                  const float* __restrict__ gw,
                                                  int* __restrict__ eids,
                                                  float* __restrict__ wts,
                                                  int N, int D) {
  __shared__ float sgw[8 * 1024];  // E=8, D<=1024
  const int tid = threadIdx.x;
  for (int i = tid; i < 8 * D; i += 256) sgw[i] = gw[i];
  __syncthreads();

  const int wv = tid >> 5, lane = tid & 31;
  const int t  = blockIdx.x * 8 + wv;
  if (t >= N) return;

  float acc[8] = {0, 0, 0, 0, 0, 0, 0, 0};
  const float* xr = x + (size_t)t * D;
  for (int d = lane; d < D; d += 32) {
    const float xv = xr[d];
#pragma unroll
    for (int e = 0; e < 8; ++e) acc[e] += xv * sgw[e * D + d];
  }
#pragma unroll
  for (int e = 0; e < 8; ++e) {
#pragma unroll
    for (int off = 16; off > 0; off >>= 1) acc[e] += __shfl_xor(acc[e], off, 32);
  }
  if (lane == 0) {
    float m = acc[0];
#pragma unroll
    for (int e = 1; e < 8; ++e) m = fmaxf(m, acc[e]);
    float p[8];
#pragma unroll
    for (int e = 0; e < 8; ++e) p[e] = __expf(acc[e] - m);
    int i1 = 0; float v1 = p[0];
#pragma unroll
    for (int e = 1; e < 8; ++e) if (p[e] > v1) { v1 = p[e]; i1 = e; }
    int i2 = -1; float v2 = -1.0f;
#pragma unroll
    for (int e = 0; e < 8; ++e) if (e != i1 && p[e] > v2) { v2 = p[e]; i2 = e; }
    const float inv = 1.0f / (v1 + v2);  // softmax denom cancels in the ratio
    eids[2 * t] = i1; eids[2 * t + 1] = i2;
    wts[2 * t]  = v1 * inv; wts[2 * t + 1] = v2 * inv;
  }
}

// ------------------------------------------------------------ ordered scan
// Single wave; ballot prefix counts give token-index-ordered positions per
// expert per k (faithful reproduction of the reference's cumsum + capacity).
__global__ void moe_scan(const int* __restrict__ eids, const float* __restrict__ wts,
                         int* __restrict__ s2t, float* __restrict__ swt,
                         int N, int cap, int rowsPerE) {
  const int lane = threadIdx.x;
  int cnt[2][8];
#pragma unroll
  for (int k = 0; k < 2; ++k)
#pragma unroll
    for (int e = 0; e < 8; ++e) cnt[k][e] = 0;

  for (int t0 = 0; t0 < N; t0 += 32) {
    const int t = t0 + lane;
    int   ek[2] = {eids[2 * t], eids[2 * t + 1]};
    float wk[2] = {wts[2 * t], wts[2 * t + 1]};
#pragma unroll
    for (int k = 0; k < 2; ++k) {
#pragma unroll
      for (int e = 0; e < 8; ++e) {
        const unsigned m = (unsigned)__ballot(ek[k] == e);
        if (ek[k] == e) {
          const int pos = cnt[k][e] + __popc(m & ((1u << lane) - 1u));
          if (pos < cap) {
            const int slot = k * cap + pos;
            s2t[e * rowsPerE + slot] = t;
            swt[e * rowsPerE + slot] = wk[k];
          }
        }
        cnt[k][e] += __popc(m);
      }
    }
  }
}

// ------------------------------------------------------------------- GEMM1
// h[e,slot,f] = silu(A @ Wg^T) * (A @ Wu^T), A rows gathered from x via slot
// map and converted fp32->bf16 into LDS. Tile: M=128 x F=64 x K=32/stage,
// wave = 16 rows x 64 cols (4 gate + 4 up accumulators).
__global__ __launch_bounds__(256) void moe_gemm1(const float* __restrict__ x,
                                                 const float* __restrict__ wig,
                                                 const float* __restrict__ wiu,
                                                 const int* __restrict__ s2t,
                                                 __bf16* __restrict__ hbuf,
                                                 int D, int F, int rowsPerE) {
  __shared__ __bf16 sA[BM * LS];
  __shared__ __bf16 sG[BN * LS];
  __shared__ __bf16 sU[BN * LS];
  __shared__ int    sTok[BM];

  const int e = blockIdx.z, row0 = blockIdx.y * BM, f0 = blockIdx.x * BN;
  const int tid = threadIdx.x, wv = tid >> 5, lane = tid & 31;

  if (tid < BM) sTok[tid] = s2t[e * rowsPerE + row0 + tid];
  __syncthreads();

  const float* wgB = wig + ((size_t)e * F + f0) * D;
  const float* wuB = wiu + ((size_t)e * F + f0) * D;

  float4 ra[4], rg[2], ru[2];
  auto loadStage = [&](int k0) {
#pragma unroll
    for (int i = 0; i < 4; ++i) {  // A: 128x32 fp32, gathered
      const int c = i * 256 + tid, r = c >> 3, c4 = (c & 7) * 4;
      const int tok = sTok[r];
      ra[i] = (tok >= 0) ? *(const float4*)(x + (size_t)tok * D + k0 + c4)
                         : make_float4(0.f, 0.f, 0.f, 0.f);
    }
#pragma unroll
    for (int i = 0; i < 2; ++i) {  // Wg/Wu: 64x32 fp32 each
      const int c = i * 256 + tid, r = c >> 3, c4 = (c & 7) * 4;
      rg[i] = *(const float4*)(wgB + (size_t)r * D + k0 + c4);
      ru[i] = *(const float4*)(wuB + (size_t)r * D + k0 + c4);
    }
  };
  auto stash = [&]() {
#pragma unroll
    for (int i = 0; i < 4; ++i) {
      const int c = i * 256 + tid, r = c >> 3, c4 = (c & 7) * 4;
      __bf16* p = sA + r * LS + c4;
      p[0] = (__bf16)ra[i].x; p[1] = (__bf16)ra[i].y;
      p[2] = (__bf16)ra[i].z; p[3] = (__bf16)ra[i].w;
    }
#pragma unroll
    for (int i = 0; i < 2; ++i) {
      const int c = i * 256 + tid, r = c >> 3, c4 = (c & 7) * 4;
      __bf16* p = sG + r * LS + c4;
      p[0] = (__bf16)rg[i].x; p[1] = (__bf16)rg[i].y;
      p[2] = (__bf16)rg[i].z; p[3] = (__bf16)rg[i].w;
      __bf16* q = sU + r * LS + c4;
      q[0] = (__bf16)ru[i].x; q[1] = (__bf16)ru[i].y;
      q[2] = (__bf16)ru[i].z; q[3] = (__bf16)ru[i].w;
    }
  };

  const v8f z8 = {0, 0, 0, 0, 0, 0, 0, 0};
  v8f accg[4] = {z8, z8, z8, z8};
  v8f accu[4] = {z8, z8, z8, z8};

  const int NS = D / BK;
  loadStage(0);
  for (int s = 0; s < NS; ++s) {
    __syncthreads();                          // prior compute done reading LDS
    stash();
    __syncthreads();                          // tile ready
    if (s + 1 < NS) loadStage((s + 1) * BK);  // global loads overlap WMMA

    // Batch fragment loads, then issue WMMAs back-to-back.
    const v16bf a = ldsFrag(sA + wv * 16 * LS, lane, LS);
    v16bf bg[4];
#pragma unroll
    for (int fs = 0; fs < 4; ++fs) bg[fs] = ldsFrag(sG + fs * 16 * LS, lane, LS);
#pragma unroll
    for (int fs = 0; fs < 4; ++fs) accg[fs] = wmma_bf16(a, bg[fs], accg[fs]);
    v16bf bu[4];
#pragma unroll
    for (int fs = 0; fs < 4; ++fs) bu[fs] = ldsFrag(sU + fs * 16 * LS, lane, LS);
#pragma unroll
    for (int fs = 0; fs < 4; ++fs) accu[fs] = wmma_bf16(a, bu[fs], accu[fs]);
  }

  // Epilogue: silu(g)*u elementwise (C layout identical for both accums).
  const int mhi = (lane & 16) ? 8 : 0, n0 = lane & 15;
#pragma unroll
  for (int fs = 0; fs < 4; ++fs) {
#pragma unroll
    for (int v = 0; v < 8; ++v) {
      const float g  = accg[fs][v];
      const float hv = (g / (1.0f + __expf(-g))) * accu[fs][v];
      const int row = row0 + wv * 16 + v + mhi;
      const int f   = f0 + fs * 16 + n0;
      hbuf[(size_t)(e * rowsPerE + row) * F + f] = (__bf16)hv;
    }
  }
}

// ------------------------------------------------------------------- GEMM2
// out[token] += w * (h @ Wo^T). A-tiles (bf16) stream via CDNA5 async
// global->LDS (double buffered, K=64 per stage = 2 WMMA k-steps); Wo
// fp32->bf16 through registers.
__global__ __launch_bounds__(256) void moe_gemm2(const __bf16* __restrict__ hbuf,
                                                 const float* __restrict__ wo,
                                                 const int* __restrict__ s2t,
                                                 const float* __restrict__ swt,
                                                 float* __restrict__ out,
                                                 int D, int F, int rowsPerE) {
  __shared__ __bf16 sA[2][BM * LS2];  // 2 x 18 KB
  __shared__ __bf16 sW[BN * LS2];     // 9 KB

  const int e = blockIdx.z, row0 = blockIdx.y * BM, d0 = blockIdx.x * BN;
  const int tid = threadIdx.x, wv = tid >> 5, lane = tid & 31;

  const float*  woB = wo + ((size_t)e * D + d0) * F;
  const __bf16* hB  = hbuf + (size_t)(e * rowsPerE + row0) * F;

  auto issueA = [&](int s) {  // 128x64 bf16 tile, 16B chunks, async -> LDS
    __bf16* dst = sA[s & 1];
    const int k0 = s * BK2;
#pragma unroll
    for (int i = 0; i < 4; ++i) {
      const int c = i * 256 + tid, r = c >> 3, ce = (c & 7) * 8;
      const __bf16* src = hB + (size_t)r * F + k0 + ce;
      __bf16* ld = dst + r * LS2 + ce;
#if MOE_ASYNC
      const unsigned           laddr = (unsigned)(size_t)ld;
      const unsigned long long gaddr = (unsigned long long)(size_t)src;
      asm volatile("global_load_async_to_lds_b128 %0, %1, off"
                   :: "v"(laddr), "v"(gaddr) : "memory");
#else
      *(uint4*)ld = *(const uint4*)src;
#endif
    }
  };

  float4 rw[4];
  auto loadW = [&](int k0) {  // Wo: 64x64 fp32
#pragma unroll
    for (int i = 0; i < 4; ++i) {
      const int c = i * 256 + tid, r = c >> 4, c4 = (c & 15) * 4;
      rw[i] = *(const float4*)(woB + (size_t)r * F + k0 + c4);
    }
  };
  auto stashW = [&]() {
#pragma unroll
    for (int i = 0; i < 4; ++i) {
      const int c = i * 256 + tid, r = c >> 4, c4 = (c & 15) * 4;
      __bf16* p = sW + r * LS2 + c4;
      p[0] = (__bf16)rw[i].x; p[1] = (__bf16)rw[i].y;
      p[2] = (__bf16)rw[i].z; p[3] = (__bf16)rw[i].w;
    }
  };

  const v8f z8 = {0, 0, 0, 0, 0, 0, 0, 0};
  v8f acc[4] = {z8, z8, z8, z8};

  const int NS = F / BK2;  // 64 stages
  issueA(0);
  loadW(0);
  for (int s = 0; s < NS; ++s) {
    __syncthreads();  // compute(s-1) done
    stashW();
#if MOE_ASYNC
    asm volatile("s_wait_asynccnt 0" ::: "memory");  // A(s) landed (per wave)
#endif
    __syncthreads();  // full tile visible
    if (s + 1 < NS) { issueA(s + 1); loadW((s + 1) * BK2); }

    const __bf16* aT = sA[s & 1] + wv * 16 * LS2;
#pragma unroll
    for (int ks = 0; ks < 2; ++ks) {  // two WMMA k-steps per stage
      const v16bf a = ldsFrag(aT + ks * 32, lane, LS2);
      v16bf b[4];
#pragma unroll
      for (int ds = 0; ds < 4; ++ds)
        b[ds] = ldsFrag(sW + ds * 16 * LS2 + ks * 32, lane, LS2);
#pragma unroll
      for (int ds = 0; ds < 4; ++ds) acc[ds] = wmma_bf16(a, b[ds], acc[ds]);
    }
  }

  // Weighted scatter back to tokens. Each output element receives exactly two
  // contributions (k=0,k=1); fl(a+b)=fl(b+a) so atomics stay deterministic.
  const int mhi = (lane & 16) ? 8 : 0, n0 = lane & 15;
#pragma unroll
  for (int v = 0; v < 8; ++v) {
    const int row = row0 + wv * 16 + v + mhi;
    const int tok = s2t[e * rowsPerE + row];
    if (tok < 0) continue;
    const float w = swt[e * rowsPerE + row];
    float* op = out + (size_t)tok * D + d0 + n0;
#pragma unroll
    for (int ds = 0; ds < 4; ++ds) atomicAdd(op + ds * 16, w * acc[ds][v]);
  }
}

// ------------------------------------------------------------------- launch
extern "C" void kernel_launch(void* const* d_in, const int* in_sizes, int n_in,
                              void* d_out, int out_size, void* d_ws, size_t ws_size,
                              hipStream_t stream) {
  const float* x   = (const float*)d_in[0];
  const float* gw  = (const float*)d_in[1];
  const float* wig = (const float*)d_in[2];
  const float* wiu = (const float*)d_in[3];
  const float* wo  = (const float*)d_in[4];
  float* out = (float*)d_out;

  const int E = 8;
  const int D = in_sizes[1] / E;           // 1024
  const int N = in_sizes[0] / D;           // 8192
  const int F = in_sizes[2] / (E * D);     // 4096
  const int cap = (N / E) * 5 / 4;         // int(N/E * 1.25) = 1280
  const int rowsPerE = 2 * cap;            // both top-k dispatches fused

  // Workspace: eids | wts | slot->token | slot weight | h (bf16, ~168 MB)
  char* w = (char*)d_ws;
  int*   eids = (int*)w;    w += (size_t)N * 2 * sizeof(int);
  float* wts  = (float*)w;  w += (size_t)N * 2 * sizeof(float);
  int*   s2t  = (int*)w;    w += (size_t)E * rowsPerE * sizeof(int);
  float* swt  = (float*)w;  w += (size_t)E * rowsPerE * sizeof(float);
  __bf16* hbuf = (__bf16*)w;

  moe_zero_out<<<2048, 256, 0, stream>>>(out, (long long)N * D);
  moe_init_slots<<<(E * rowsPerE + 255) / 256, 256, 0, stream>>>(s2t, swt,
                                                                 E * rowsPerE);
  moe_router<<<N / 8, 256, 0, stream>>>(x, gw, eids, wts, N, D);
  moe_scan<<<1, 32, 0, stream>>>(eids, wts, s2t, swt, N, cap, rowsPerE);

  dim3 g1(F / BN, rowsPerE / BM, E);       // (64, 20, 8)
  moe_gemm1<<<g1, 256, 0, stream>>>(x, wig, wiu, s2t, hbuf, D, F, rowsPerE);

  dim3 g2(D / BN, rowsPerE / BM, E);       // (16, 20, 8)
  moe_gemm2<<<g2, 256, 0, stream>>>(hbuf, wo, s2t, swt, out, D, F, rowsPerE);
}
